// MoEFFN_27487790694795
// MI455X (gfx1250) — matrix operand order repaired and checked by reading, not compiled, not run on hardware
//
#include <hip/hip_runtime.h>
#include <cstdint>
#include <cstddef>

// ---------------------------------------------------------------------------
// MoE FFN (top-2 of 8 experts), D=768, H=3072, T=1024, f32 in/out.
// MI455X (gfx1250) strategy:
//  * Routed grouped GEMM (19.3 GFLOP) instead of dense 8-expert (77 GFLOP).
//  * One streaming pre-pass converts f32 weights -> bf16, TRANSPOSED to
//    [N][K], plus x -> bf16. Cost: 151 MB read + 75.5 MB write ~ 9.7us at
//    23.3 TB/s. The bf16 copies (75.5 MB) fit in the 192 MB L2, so the GEMM
//    passes never re-touch HBM for weights.
//  * GEMM hot loops are pure b128 loads + v_wmma_f32_16x16x32_bf16 (f32
//    accum). No per-use f32->bf16 packing (round-1 histogram showed that
//    VALU packing dominated: valu=2069 vs wmma=8).
//  * 64x64 wave tile (16 v8f accumulators, ~200 VGPRs of the 1024 budget):
//    4 A fragments x 4 B fragments -> 16 WMMAs per 32-deep K step, cutting
//    L2 weight re-reads to ~300 MB total.
//  * h intermediate stored bf16 (feeds GEMM2's A operand directly).
//  * Deterministic slot-based combine (no output atomics).
// ---------------------------------------------------------------------------

typedef __attribute__((ext_vector_type(16))) __bf16 v16bf;
typedef __attribute__((ext_vector_type(8)))  float  v8f;

constexpr int DIM  = 768;
constexpr int NEXP = 8;
constexpr int HID  = 3072;
constexpr int NTOK = 1024;
constexpr int NSLOT = NTOK * 2;          // exactly T*TOP_K routed assignments

// Workspace layout (bytes). Total ~96 MB.
constexpr size_t WS_COUNTS   = 0;                               // int[8]
constexpr size_t WS_OFFSETS  = 64;                              // int[8]
constexpr size_t WS_CURSOR   = 128;                             // int[8]
constexpr size_t WS_TOKEXP   = 256;                             // int[2048]
constexpr size_t WS_TOKSCORE = WS_TOKEXP   + (size_t)NSLOT * 4; // float[2048]
constexpr size_t WS_TOKSLOT  = WS_TOKSCORE + (size_t)NSLOT * 4; // int[2048]
constexpr size_t WS_SLOTTOK  = WS_TOKSLOT  + (size_t)NSLOT * 4; // int[2048]
constexpr size_t WS_H        = ((WS_SLOTTOK + (size_t)NSLOT * 4 + 255) / 256) * 256; // bf16[2048*3072]
constexpr size_t WS_Y        = WS_H   + (size_t)NSLOT * HID * 2;        // f32 [2048*768]
constexpr size_t WS_XB       = WS_Y   + (size_t)NSLOT * DIM * 4;        // bf16[1024*768]
constexpr size_t WS_W1T      = WS_XB  + (size_t)NTOK * DIM * 2;         // bf16[8*3072*768]  ([e][n][k])
constexpr size_t WS_W2T      = WS_W1T + (size_t)NEXP * HID * DIM * 2;   // bf16[8*768*3072]  ([e][n][k])

// --- helpers ---------------------------------------------------------------

__device__ __forceinline__ unsigned pack_bf16(float a, float b) {
  // round-to-nearest-even f32 -> bf16, packed (a low half, b high half)
  unsigned ua = __float_as_uint(a), ub = __float_as_uint(b);
  ua += 0x7FFFu + ((ua >> 16) & 1u);
  ub += 0x7FFFu + ((ub >> 16) & 1u);
  return (ua >> 16) | (ub & 0xFFFF0000u);
}

__device__ __forceinline__ unsigned short bf16_of(float a) {
  unsigned ua = __float_as_uint(a);
  ua += 0x7FFFu + ((ua >> 16) & 1u);
  return (unsigned short)(ua >> 16);
}

__device__ __forceinline__ float gelu_exact(float x) {
  return 0.5f * x * (1.0f + erff(x * 0.70710678118654752f));
}

union Frag { unsigned u[8]; v16bf v; };

__device__ __forceinline__ void load_frag_bf16(Frag& f, const unsigned short* p,
                                               int off_lo, int off_hi) {
  // elements 0..7 at p+off_lo (contig bf16), elements 8..15 at p+off_hi
  uint4 q0 = *(const uint4*)(p + off_lo);
  uint4 q1 = *(const uint4*)(p + off_hi);
  f.u[0] = q0.x; f.u[1] = q0.y; f.u[2] = q0.z; f.u[3] = q0.w;
  f.u[4] = q1.x; f.u[5] = q1.y; f.u[6] = q1.z; f.u[7] = q1.w;
}

// --- weight pre-conversion -------------------------------------------------
// src: f32 [E][K][N] row-major -> dst: bf16 [E][N][K] (transposed), via LDS
// tile so both global reads and writes are coalesced.

__global__ __launch_bounds__(256) void convert_transpose_bf16(
    const float* __restrict__ src, unsigned short* __restrict__ dst,
    int K, int N) {
  __shared__ unsigned short tile[64 * 66];   // [n][k], stride 66 (33 dwords,
                                             // odd -> conflict-free columns)
  int e = blockIdx.z;
  src += (size_t)e * K * N;
  dst += (size_t)e * K * N;
  int nBase = blockIdx.x * 64, kBase = blockIdx.y * 64;

  int tx = threadIdx.x & 63;     // n within tile
  int ty = threadIdx.x >> 6;     // 0..3
#pragma unroll
  for (int kk = ty; kk < 64; kk += 4)
    tile[tx * 66 + kk] = bf16_of(src[(size_t)(kBase + kk) * N + nBase + tx]);
  __syncthreads();

  int n = threadIdx.x >> 2;      // 0..63
  int c = threadIdx.x & 3;       // 16-element k-chunk
  const unsigned* lp = (const unsigned*)tile + n * 33 + c * 8;
  uint4 a = { lp[0], lp[1], lp[2], lp[3] };
  uint4 b = { lp[4], lp[5], lp[6], lp[7] };
  uint4* op = (uint4*)(dst + (size_t)(nBase + n) * K + kBase + c * 16);
  op[0] = a; op[1] = b;
}

__global__ __launch_bounds__(256) void convert_x_bf16(
    const float* __restrict__ x, unsigned short* __restrict__ xb) {
  int i = (blockIdx.x * 256 + threadIdx.x) * 8;   // NTOK*DIM total elements
  float4 f0 = *(const float4*)(x + i);
  float4 f1 = *(const float4*)(x + i + 4);
  uint4 q = { pack_bf16(f0.x, f0.y), pack_bf16(f0.z, f0.w),
              pack_bf16(f1.x, f1.y), pack_bf16(f1.z, f1.w) };
  *(uint4*)(xb + i) = q;
}

// --- routing kernels -------------------------------------------------------

__global__ void zero_small(unsigned char* ws) {
  int i = threadIdx.x;
  if (i < NEXP) {
    ((int*)(ws + WS_COUNTS))[i] = 0;
    ((int*)(ws + WS_CURSOR))[i] = 0;
  }
}

__global__ __launch_bounds__(256) void gate_topk(
    const float* __restrict__ x, const float* __restrict__ Wg,
    const float* __restrict__ bg, unsigned char* __restrict__ ws) {
  int t = blockIdx.x * blockDim.x + threadIdx.x;
  if (t >= NTOK) return;
  int*   counts   = (int*)(ws + WS_COUNTS);
  int*   tokExp   = (int*)(ws + WS_TOKEXP);
  float* tokScore = (float*)(ws + WS_TOKSCORE);

  const float* xr = x + (size_t)t * DIM;
  float lg[NEXP];
#pragma unroll
  for (int e = 0; e < NEXP; e++) lg[e] = bg[e];
  for (int k = 0; k < DIM; k++) {
    float xv = xr[k];
    const float4* wr = (const float4*)(Wg + (size_t)k * NEXP);
    float4 w0 = wr[0], w1 = wr[1];
    lg[0] += xv * w0.x; lg[1] += xv * w0.y; lg[2] += xv * w0.z; lg[3] += xv * w0.w;
    lg[4] += xv * w1.x; lg[5] += xv * w1.y; lg[6] += xv * w1.z; lg[7] += xv * w1.w;
  }
  // stable top-2 (strict > keeps lowest index on ties, matching lax.top_k)
  int e0 = 0; float s0 = lg[0];
#pragma unroll
  for (int e = 1; e < NEXP; e++) if (lg[e] > s0) { s0 = lg[e]; e0 = e; }
  int e1 = -1; float s1 = -3.402823466e+38f;
#pragma unroll
  for (int e = 0; e < NEXP; e++) {
    if (e == e0) continue;
    if (lg[e] > s1) { s1 = lg[e]; e1 = e; }
  }
  tokExp[2 * t] = e0;  tokExp[2 * t + 1] = e1;
  tokScore[2 * t] = s0; tokScore[2 * t + 1] = s1;
  atomicAdd(&counts[e0], 1);
  atomicAdd(&counts[e1], 1);
}

__global__ void prefix_scan(unsigned char* ws) {
  int* counts = (int*)(ws + WS_COUNTS);
  int* offs   = (int*)(ws + WS_OFFSETS);
  int a = 0;
  for (int e = 0; e < NEXP; e++) { offs[e] = a; a += counts[e]; }
}

__global__ __launch_bounds__(256) void fill_slots(unsigned char* ws) {
  int t = blockIdx.x * blockDim.x + threadIdx.x;
  if (t >= NTOK) return;
  int* cursor  = (int*)(ws + WS_CURSOR);
  int* offs    = (int*)(ws + WS_OFFSETS);
  int* tokExp  = (int*)(ws + WS_TOKEXP);
  int* tokSlot = (int*)(ws + WS_TOKSLOT);
  int* slotTok = (int*)(ws + WS_SLOTTOK);
#pragma unroll
  for (int k = 0; k < 2; k++) {
    int e = tokExp[2 * t + k];
    int pos = atomicAdd(&cursor[e], 1);
    int slot = offs[e] + pos;
    slotTok[slot] = t;
    tokSlot[2 * t + k] = slot;
  }
}

// --- GEMM1: h = gelu(x_gathered @ W1[e] + b1[e]), stored bf16 ---------------
// One wave computes a 64x64 tile: 4 A fragments x 4 B fragments = 16 WMMAs
// per 32-deep K step. grid = (HID/64, NEXP*16); early-exit on counts[e].

__global__ __launch_bounds__(32) void gemm1_gelu(
    const unsigned short* __restrict__ XB, const unsigned short* __restrict__ W1T,
    const float* __restrict__ b1, unsigned char* __restrict__ ws) {
  const int* counts  = (const int*)(ws + WS_COUNTS);
  const int* offs    = (const int*)(ws + WS_OFFSETS);
  const int* slotTok = (const int*)(ws + WS_SLOTTOK);
  unsigned short* H  = (unsigned short*)(ws + WS_H);

  int e  = blockIdx.y >> 4;
  int mt = blockIdx.y & 15;
  int cnt = counts[e];
  if (mt * 64 >= cnt) return;
  int segBase = offs[e];
  int nBase = blockIdx.x * 64;

  int lane  = threadIdx.x & 31;
  int lmod  = lane & 15;
  int lhalf = lane >> 4;
  const int kbo  = lhalf * 16;   // A: K = 16*half + i
  const int kboB = lhalf * 16;   // B: K = 16*half + i

  // A row pointers (gathered tokens); clamp OOB rows to a valid slot
  const unsigned short* ap[4];
#pragma unroll
  for (int mi = 0; mi < 4; mi++) {
    int mloc = mt * 64 + mi * 16 + lmod;
    int slotA = segBase + (mloc < cnt ? mloc : mt * 64);
    ap[mi] = XB + (size_t)slotTok[slotA] * DIM + kbo;
  }
  // B column pointers (W1T is [e][n][k], K contiguous)
  const unsigned short* bp[4];
#pragma unroll
  for (int j = 0; j < 4; j++)
    bp[j] = W1T + ((size_t)e * HID + nBase + 16 * j + lmod) * DIM + kboB;

  v8f acc[16] = {};
  for (int kk = 0; kk < DIM; kk += 32) {
    Frag Af[4];
#pragma unroll
    for (int mi = 0; mi < 4; mi++) load_frag_bf16(Af[mi], ap[mi], kk, kk + 8);
#pragma unroll
    for (int j = 0; j < 4; j++) {
      Frag B;
      load_frag_bf16(B, bp[j], kk, kk + 8);
#pragma unroll
      for (int mi = 0; mi < 4; mi++)
        acc[mi * 4 + j] = __builtin_amdgcn_wmma_f32_16x16x32_bf16(
            false, Af[mi].v, false, B.v, (short)0, acc[mi * 4 + j], false, false);
    }
  }

  const float* b1e = b1 + (size_t)e * HID;
#pragma unroll
  for (int j = 0; j < 4; j++) {
    int n = nBase + 16 * j + lmod;
    float bias = b1e[n];
#pragma unroll
    for (int mi = 0; mi < 4; mi++) {
#pragma unroll
      for (int r = 0; r < 8; r++) {
        int ml = mt * 64 + mi * 16 + r + lhalf * 8;  // C/D: row = reg + 8*half
        if (ml < cnt) {
          float g = gelu_exact(acc[mi * 4 + j][r] + bias);
          H[(size_t)(segBase + ml) * HID + n] = bf16_of(g);
        }
      }
    }
  }
}

// --- GEMM2: y = h @ W2[e] + b2[e], stored f32 per slot ----------------------
// grid = (DIM/64, NEXP*16)

__global__ __launch_bounds__(32) void gemm2_bias(
    const unsigned short* __restrict__ W2T, const float* __restrict__ b2,
    unsigned char* __restrict__ ws) {
  const int* counts = (const int*)(ws + WS_COUNTS);
  const int* offs   = (const int*)(ws + WS_OFFSETS);
  const unsigned short* H = (const unsigned short*)(ws + WS_H);
  float* Y = (float*)(ws + WS_Y);

  int e  = blockIdx.y >> 4;
  int mt = blockIdx.y & 15;
  int cnt = counts[e];
  if (mt * 64 >= cnt) return;
  int segBase = offs[e];
  int nBase = blockIdx.x * 64;

  int lane  = threadIdx.x & 31;
  int lmod  = lane & 15;
  int lhalf = lane >> 4;
  const int kbo  = lhalf * 16;
  const int kboB = lhalf * 16;

  const unsigned short* ap[4];
#pragma unroll
  for (int mi = 0; mi < 4; mi++) {
    int mloc = mt * 64 + mi * 16 + lmod;
    int slotA = segBase + (mloc < cnt ? mloc : mt * 64);
    ap[mi] = H + (size_t)slotA * HID + kbo;
  }
  const unsigned short* bp[4];
#pragma unroll
  for (int j = 0; j < 4; j++)
    bp[j] = W2T + ((size_t)e * DIM + nBase + 16 * j + lmod) * HID + kboB;

  v8f acc[16] = {};
  for (int kk = 0; kk < HID; kk += 32) {
    Frag Af[4];
#pragma unroll
    for (int mi = 0; mi < 4; mi++) load_frag_bf16(Af[mi], ap[mi], kk, kk + 8);
#pragma unroll
    for (int j = 0; j < 4; j++) {
      Frag B;
      load_frag_bf16(B, bp[j], kk, kk + 8);
#pragma unroll
      for (int mi = 0; mi < 4; mi++)
        acc[mi * 4 + j] = __builtin_amdgcn_wmma_f32_16x16x32_bf16(
            false, Af[mi].v, false, B.v, (short)0, acc[mi * 4 + j], false, false);
    }
  }

  const float* b2e = b2 + (size_t)e * DIM;
#pragma unroll
  for (int j = 0; j < 4; j++) {
    int n = nBase + 16 * j + lmod;
    float bias = b2e[n];
#pragma unroll
    for (int mi = 0; mi < 4; mi++) {
#pragma unroll
      for (int r = 0; r < 8; r++) {
        int ml = mt * 64 + mi * 16 + r + lhalf * 8;
        if (ml < cnt)
          Y[(size_t)(segBase + ml) * DIM + n] = acc[mi * 4 + j][r] + bias;
      }
    }
  }
}

// --- combine: out[t] = s0*y[slot0] + s1*y[slot1] ---------------------------

__global__ __launch_bounds__(256) void combine(
    const unsigned char* __restrict__ ws, float* __restrict__ out) {
  int t = blockIdx.x;
  const int*   tokSlot  = (const int*)(ws + WS_TOKSLOT);
  const float* tokScore = (const float*)(ws + WS_TOKSCORE);
  const float* Y = (const float*)(ws + WS_Y);
  int   sA = tokSlot[2 * t],   sB = tokSlot[2 * t + 1];
  float wA = tokScore[2 * t],  wB = tokScore[2 * t + 1];
  const float* yA = Y + (size_t)sA * DIM;
  const float* yB = Y + (size_t)sB * DIM;
  for (int d = threadIdx.x; d < DIM; d += blockDim.x)
    out[(size_t)t * DIM + d] = wA * yA[d] + wB * yB[d];
}

// --- launch ----------------------------------------------------------------

extern "C" void kernel_launch(void* const* d_in, const int* in_sizes, int n_in,
                              void* d_out, int out_size, void* d_ws, size_t ws_size,
                              hipStream_t stream) {
  const float* x  = (const float*)d_in[0];
  const float* Wg = (const float*)d_in[1];
  const float* bg = (const float*)d_in[2];
  const float* W1 = (const float*)d_in[3];
  const float* b1 = (const float*)d_in[4];
  const float* W2 = (const float*)d_in[5];
  const float* b2 = (const float*)d_in[6];
  float* out = (float*)d_out;
  unsigned char* ws = (unsigned char*)d_ws;
  (void)in_sizes; (void)n_in; (void)out_size; (void)ws_size;

  unsigned short* XB  = (unsigned short*)(ws + WS_XB);
  unsigned short* W1T = (unsigned short*)(ws + WS_W1T);
  unsigned short* W2T = (unsigned short*)(ws + WS_W2T);

  // pre-pass: bf16 conversion (+ weight transpose to [e][n][k])
  convert_x_bf16<<<dim3(NTOK * DIM / (256 * 8)), 256, 0, stream>>>(x, XB);
  convert_transpose_bf16<<<dim3(HID / 64, DIM / 64, NEXP), 256, 0, stream>>>(
      W1, W1T, DIM, HID);
  convert_transpose_bf16<<<dim3(DIM / 64, HID / 64, NEXP), 256, 0, stream>>>(
      W2, W2T, HID, DIM);

  // routing
  zero_small<<<1, 32, 0, stream>>>(ws);
  gate_topk<<<dim3((NTOK + 255) / 256), 256, 0, stream>>>(x, Wg, bg, ws);
  prefix_scan<<<1, 1, 0, stream>>>(ws);
  fill_slots<<<dim3((NTOK + 255) / 256), 256, 0, stream>>>(ws);

  // grouped GEMMs
  gemm1_gelu<<<dim3(HID / 64, NEXP * 16), 32, 0, stream>>>(XB, W1T, b1, ws);
  gemm2_bias<<<dim3(DIM / 64, NEXP * 16), 32, 0, stream>>>(W2T, b2, ws);

  // combine
  combine<<<dim3(NTOK), 256, 0, stream>>>(ws, out);
}
